// BLHmmLm_56435870270167
// MI455X (gfx1250) — matrix operand order repaired and checked by reading, not compiled, not run on hardware
//
#include <hip/hip_runtime.h>

// ---------------- types ----------------
typedef __bf16 bf16;
typedef bf16  bf16x8  __attribute__((ext_vector_type(8)));
typedef bf16  bf16x16 __attribute__((ext_vector_type(16)));
typedef float f32x8   __attribute__((ext_vector_type(8)));

union Frag {
  bf16x16 v;
  struct { bf16x8 lo, hi; } h;
};

static __device__ __forceinline__ f32x8 wmma_bf16(const Frag& a, const Frag& b, f32x8 c) {
  // D = A(16x32 bf16) x B(32x16 bf16) + C(16x16 f32)
  return __builtin_amdgcn_wmma_f32_16x16x32_bf16(false, a.v, false, b.v, (short)0, c,
                                                 false, false);
}

// ---------------- problem dims ----------------
constexpr int Cn = 8192;   // states
constexpr int Hn = 256;    // hidden
constexpr int Dn = 512;    // features
constexpr int Vn = 10000;  // vocab
constexpr int Nn_ = 16;    // batch
constexpr int Tn = 256;    // time
constexpr int NT = Nn_ * Tn;

// ---------------- small elementwise / reduction kernels ----------------
__global__ void k_cvt_bf16(const float* __restrict__ in, bf16* __restrict__ out, long n) {
  long i = (long)blockIdx.x * blockDim.x + threadIdx.x;
  if (i < n) out[i] = (bf16)in[i];
}

__global__ void k_tcvt(const float* __restrict__ in, bf16* __restrict__ out, int R, int Cc) {
  long i = (long)blockIdx.x * blockDim.x + threadIdx.x;
  if (i < (long)R * Cc) {
    int r = (int)(i / Cc), c = (int)(i % Cc);
    out[(size_t)c * R + r] = (bf16)in[i];
  }
}

__global__ __launch_bounds__(256) void k_rowsq_f32(const float* __restrict__ X,
                                                   float* __restrict__ sq, int cols) {
  __shared__ float red[256];
  int row = blockIdx.x, tid = threadIdx.x;
  float s = 0.f;
  for (int c = tid; c < cols; c += 256) { float v = X[(size_t)row * cols + c]; s += v * v; }
  red[tid] = s; __syncthreads();
  for (int st = 128; st > 0; st >>= 1) { if (tid < st) red[tid] += red[tid + st]; __syncthreads(); }
  if (tid == 0) sq[row] = red[0];
}

__global__ __launch_bounds__(256) void k_rowsq_bf16(const bf16* __restrict__ X,
                                                    float* __restrict__ sq, int cols) {
  __shared__ float red[256];
  int row = blockIdx.x, tid = threadIdx.x;
  float s = 0.f;
  for (int c = tid; c < cols; c += 256) { float v = (float)X[(size_t)row * cols + c]; s += v * v; }
  red[tid] = s; __syncthreads();
  for (int st = 128; st > 0; st >>= 1) { if (tid < st) red[tid] += red[tid + st]; __syncthreads(); }
  if (tid == 0) sq[row] = red[0];
}

// row-wise max over D, optionally write E = exp(L - rmax) as bf16
__global__ __launch_bounds__(256) void k_rowmax_exp(const float* __restrict__ L,
                                                    float* __restrict__ rmax,
                                                    bf16* __restrict__ E, int Dc) {
  __shared__ float red[256];
  int row = blockIdx.x, tid = threadIdx.x;
  float m = -3.0e38f;
  for (int d = tid; d < Dc; d += 256) m = fmaxf(m, L[(size_t)row * Dc + d]);
  red[tid] = m; __syncthreads();
  for (int st = 128; st > 0; st >>= 1) { if (tid < st) red[tid] = fmaxf(red[tid], red[tid + st]); __syncthreads(); }
  m = red[0]; __syncthreads();
  if (tid == 0) rmax[row] = m;
  if (E)
    for (int d = tid; d < Dc; d += 256)
      E[(size_t)row * Dc + d] = (bf16)expf(L[(size_t)row * Dc + d] - m);
}

// per-feature logsumexp over rows: out[f] = log sum_r exp(L[r,f])
__global__ __launch_bounds__(256) void k_collse(const float* __restrict__ L,
                                                float* __restrict__ out, int rows, int Dc) {
  __shared__ float red[256];
  int f = blockIdx.x, tid = threadIdx.x;
  float m = -3.0e38f;
  for (int r = tid; r < rows; r += 256) m = fmaxf(m, L[(size_t)r * Dc + f]);
  red[tid] = m; __syncthreads();
  for (int st = 128; st > 0; st >>= 1) { if (tid < st) red[tid] = fmaxf(red[tid], red[tid + st]); __syncthreads(); }
  m = red[0]; __syncthreads();
  float s = 0.f;
  for (int r = tid; r < rows; r += 256) s += expf(L[(size_t)r * Dc + f] - m);
  red[tid] = s; __syncthreads();
  for (int st = 128; st > 0; st >>= 1) { if (tid < st) red[tid] += red[tid + st]; __syncthreads(); }
  if (tid == 0) out[f] = m + logf(red[0]);
}

// logden[i] = logsumexp_f( L[i,f] - rmax[i] + lw[f] )
__global__ __launch_bounds__(256) void k_logden(const float* __restrict__ L,
                                                const float* __restrict__ rmax,
                                                const float* __restrict__ lw,
                                                float* __restrict__ ld, int Dc) {
  __shared__ float red[256];
  int row = blockIdx.x, tid = threadIdx.x;
  float rm = rmax[row];
  float m = -3.0e38f;
  for (int f = tid; f < Dc; f += 256) m = fmaxf(m, L[(size_t)row * Dc + f] - rm + lw[f]);
  red[tid] = m; __syncthreads();
  for (int st = 128; st > 0; st >>= 1) { if (tid < st) red[tid] = fmaxf(red[tid], red[tid + st]); __syncthreads(); }
  m = red[0]; __syncthreads();
  float s = 0.f;
  for (int f = tid; f < Dc; f += 256) s += expf(L[(size_t)row * Dc + f] - rm + lw[f] - m);
  red[tid] = s; __syncthreads();
  for (int st = 128; st > 0; st >>= 1) { if (tid < st) red[tid] += red[tid + st]; __syncthreads(); }
  if (tid == 0) ld[row] = m + logf(red[0]);
}

// Bg[tok,f] = exp(lt[text[tok],f] - bmax_t[text[tok]]); bmax_tok[tok] = bmax_t[text[tok]]
__global__ void k_gather(const int* __restrict__ text, const float* __restrict__ lt,
                         const float* __restrict__ bmax_t, bf16* __restrict__ Bg,
                         float* __restrict__ bmax_tok) {
  long i = (long)blockIdx.x * blockDim.x + threadIdx.x;
  if (i >= (long)NT * Dn) return;
  int tok = (int)(i >> 9), f = (int)(i & 511);
  int v = text[tok];
  float bm = bmax_t[v];
  Bg[i] = (bf16)expf(lt[(size_t)v * Dn + f] - bm);
  if (f == 0) bmax_tok[tok] = bm;
}

// start_mlp chain on a single [H] vector + projection (tiny; one block)
__global__ __launch_bounds__(256) void k_start_chain(
    const float* semb, const float* smw, const float* smb,
    const float* w11, const float* b11, const float* w12, const float* b12,
    const float* w21, const float* b21, const float* w22, const float* b22,
    const float* proj, float* A0) {
  __shared__ float xa[256], xb[256], hh[256], tm[256], l[512], red[256];
  int tid = threadIdx.x;
  xa[tid] = semb[tid]; __syncthreads();
  { float a = smb[tid]; for (int k = 0; k < 256; ++k) a += xa[k] * smw[k * 256 + tid]; xb[tid] = a; } __syncthreads();
  { float a = b11[tid]; for (int k = 0; k < 256; ++k) a += xb[k] * w11[k * 256 + tid]; hh[tid] = fmaxf(a, 0.f); } __syncthreads();
  { float a = b12[tid]; for (int k = 0; k < 256; ++k) a += hh[k] * w12[k * 256 + tid]; tm[tid] = fmaxf(a, 0.f); } __syncthreads();
  xa[tid] = xb[tid] + tm[tid]; __syncthreads();
  { float a = b21[tid]; for (int k = 0; k < 256; ++k) a += xa[k] * w21[k * 256 + tid]; hh[tid] = fmaxf(a, 0.f); } __syncthreads();
  { float a = b22[tid]; for (int k = 0; k < 256; ++k) a += hh[k] * w22[k * 256 + tid]; tm[tid] = fmaxf(a, 0.f); } __syncthreads();
  xb[tid] = xa[tid] + tm[tid]; __syncthreads();
  red[tid] = xb[tid] * xb[tid]; __syncthreads();
  for (int st = 128; st > 0; st >>= 1) { if (tid < st) red[tid] += red[tid + st]; __syncthreads(); }
  float xsq = red[0]; __syncthreads();
  for (int d = tid; d < 512; d += 256) {
    float a = 0.f; for (int k = 0; k < 256; ++k) a += xb[k] * proj[k * 512 + d];
    l[d] = a - 0.5f * xsq;
  } __syncthreads();
  red[tid] = fmaxf(l[tid], l[tid + 256]); __syncthreads();
  for (int st = 128; st > 0; st >>= 1) { if (tid < st) red[tid] = fmaxf(red[tid], red[tid + st]); __syncthreads(); }
  float amax = red[0]; __syncthreads();
  for (int d = tid; d < 512; d += 256) A0[d] = expf(l[d] - amax);
}

__global__ __launch_bounds__(256) void k_start_gemv(const float* __restrict__ A0,
                                                    const bf16* __restrict__ By,
                                                    const float* __restrict__ bmax_y,
                                                    float* __restrict__ s) {
  int c = blockIdx.x * 256 + threadIdx.x;
  float acc = 0.f;
  const bf16* row = By + (size_t)c * Dn;
  for (int f = 0; f < Dn; ++f) acc += A0[f] * (float)row[f];
  s[c] = logf(fmaxf(acc, 1e-37f)) + bmax_y[c];
}

__global__ __launch_bounds__(256) void k_lse_vec(const float* __restrict__ s, int n,
                                                 float* __restrict__ out) {
  __shared__ float red[256];
  int tid = threadIdx.x;
  float m = -3.0e38f;
  for (int i = tid; i < n; i += 256) m = fmaxf(m, s[i]);
  red[tid] = m; __syncthreads();
  for (int st = 128; st > 0; st >>= 1) { if (tid < st) red[tid] = fmaxf(red[tid], red[tid + st]); __syncthreads(); }
  m = red[0]; __syncthreads();
  float sm = 0.f;
  for (int i = tid; i < n; i += 256) sm += expf(s[i] - m);
  red[tid] = sm; __syncthreads();
  for (int st = 128; st > 0; st >>= 1) { if (tid < st) red[tid] += red[tid + st]; __syncthreads(); }
  if (tid == 0) out[0] = m + logf(red[0]);
}

__global__ void k_sub(const float* __restrict__ s, const float* __restrict__ lse,
                      float* __restrict__ out, int n) {
  int i = blockIdx.x * blockDim.x + threadIdx.x;
  if (i < n) out[i] = s[i] - lse[0];
}

__global__ void k_init_au(const float* __restrict__ startl, const bf16* __restrict__ pe0,
                          float* __restrict__ au) {
  long i = (long)blockIdx.x * blockDim.x + threadIdx.x;
  if (i < (long)Nn_ * Cn) au[i] = startl[i & (Cn - 1)] + (float)pe0[i];
}

// ---------------- main tiled WMMA bf16 NT GEMM ----------------
// C(MxN) = A(MxK) * B(NxK)^T   (both bf16 row-major, lda = ldb = K), f32 accum.
// Block tile 128x128, 8 waves (2x4), wave tile 64x32, K-step 32.
struct GArgs {
  const bf16* A; const bf16* B;
  int M, Nc, K;
  float* outF; bf16* outB;
  const float* vrow; const float* vcol; const bf16* res;
};

// EPI: 0 relu(acc+bias[col]) -> bf16
//      1 res + relu(acc+bias[col]) -> bf16
//      2 acc - 0.5*vrow[row] -> f32        (FAVOR+ feature logits)
//      3 acc * exp(vrow[row]-vcol[col]) -> bf16   (trans_exp, already transposed)
//      4 log(acc) + vrow[row] - vcol[col] -> bf16 at permuted [t][n][c]  (p_emit)
template <int EPI>
__global__ __launch_bounds__(256) void k_gemm_nt(GArgs g) {
  __shared__ __align__(16) bf16 As[128 * 32];
  __shared__ __align__(16) bf16 Bs[128 * 32];
  const int tid = threadIdx.x;
  const int lane = tid & 31, wid = tid >> 5;
  const int waveM = wid >> 2, waveN = wid & 3;
  const int m16 = lane & 15, hi16 = lane >> 4;
  const int bm = blockIdx.y * 128, bn = blockIdx.x * 128;
  f32x8 acc[4][2] = {};

  for (int k0 = 0; k0 < g.K; k0 += 32) {
    __syncthreads();
#pragma unroll
    for (int i = 0; i < 2; ++i) {
      int chunk = i * 256 + tid;
      int row = chunk >> 2, sub = chunk & 3;
      int ga = bm + row; if (ga > g.M - 1) ga = g.M - 1;
      *(bf16x8*)(As + row * 32 + sub * 8) =
          *(const bf16x8*)(g.A + (size_t)ga * g.K + k0 + sub * 8);
      int gb = bn + row; if (gb > g.Nc - 1) gb = g.Nc - 1;
      *(bf16x8*)(Bs + row * 32 + sub * 8) =
          *(const bf16x8*)(g.B + (size_t)gb * g.K + k0 + sub * 8);
    }
    __syncthreads();
    Frag a[4], b[2];
#pragma unroll
    for (int mi = 0; mi < 4; ++mi) {
      const bf16* p = As + (waveM * 64 + mi * 16 + m16) * 32;
      a[mi].h.lo = *(const bf16x8*)(p + hi16 * 8);
      a[mi].h.hi = *(const bf16x8*)(p + 16 + hi16 * 8);
    }
#pragma unroll
    for (int ni = 0; ni < 2; ++ni) {
      const bf16* p = Bs + (waveN * 32 + ni * 16 + m16) * 32;
      b[ni].h.lo = *(const bf16x8*)(p + hi16 * 8);
      b[ni].h.hi = *(const bf16x8*)(p + 16 + hi16 * 8);
    }
#pragma unroll
    for (int mi = 0; mi < 4; ++mi)
#pragma unroll
      for (int ni = 0; ni < 2; ++ni)
        acc[mi][ni] = wmma_bf16(a[mi], b[ni], acc[mi][ni]);
  }

#pragma unroll
  for (int mi = 0; mi < 4; ++mi)
#pragma unroll
    for (int ni = 0; ni < 2; ++ni) {
      int col = bn + waveN * 32 + ni * 16 + m16;
#pragma unroll
      for (int r = 0; r < 8; ++r) {
        int row = bm + waveM * 64 + mi * 16 + hi16 * 8 + r;
        if (row >= g.M || col >= g.Nc) continue;
        float v = acc[mi][ni][r];
        size_t idx = (size_t)row * g.Nc + col;
        if (EPI == 0) {
          g.outB[idx] = (bf16)fmaxf(v + g.vcol[col], 0.f);
        } else if (EPI == 1) {
          g.outB[idx] = (bf16)((float)g.res[idx] + fmaxf(v + g.vcol[col], 0.f));
        } else if (EPI == 2) {
          g.outF[idx] = v - 0.5f * g.vrow[row];
        } else if (EPI == 3) {
          g.outB[idx] = (bf16)(v * expf(g.vrow[row] - g.vcol[col]));
        } else {
          int tt = row & (Tn - 1), nb = row >> 8;  // tok = n*T + t
          size_t pidx = ((size_t)tt * Nn_ + nb) * Cn + col;
          g.outB[pidx] = (bf16)(logf(fmaxf(v, 1e-37f)) + g.vrow[row] - g.vcol[col]);
        }
      }
    }
}

// ---------------- scan step: au[16,C] = log(alpha[16,C] @ transT^T) + pe_t ----------------
// transT is stored [c][k] row-major so B fragments are two contiguous 16B runs per lane (L2-resident bf16).
__global__ __launch_bounds__(256) void k_scan_gemm(const bf16* __restrict__ alp,
                                                   const bf16* __restrict__ trT,
                                                   const bf16* __restrict__ pet,
                                                   float* __restrict__ au) {
  const int tid = threadIdx.x, lane = tid & 31, wid = tid >> 5;
  const int m16 = lane & 15, hi16 = lane >> 4;
  const int colBase = blockIdx.x * 256 + wid * 32;
  f32x8 acc0 = {}, acc1 = {};
  const bf16* arow = alp + (size_t)m16 * Cn;
  const bf16* b0row = trT + (size_t)(colBase + m16) * Cn;
  const bf16* b1row = trT + (size_t)(colBase + 16 + m16) * Cn;
  for (int k0 = 0; k0 < Cn; k0 += 32) {
    Frag a, b0, b1;
    a.h.lo = *(const bf16x8*)(arow + k0 + hi16 * 8);
    a.h.hi = *(const bf16x8*)(arow + k0 + 16 + hi16 * 8);
    b0.h.lo = *(const bf16x8*)(b0row + k0 + hi16 * 8);
    b0.h.hi = *(const bf16x8*)(b0row + k0 + 16 + hi16 * 8);
    b1.h.lo = *(const bf16x8*)(b1row + k0 + hi16 * 8);
    b1.h.hi = *(const bf16x8*)(b1row + k0 + 16 + hi16 * 8);
    if (k0 + 32 < Cn) {
      __builtin_prefetch(b0row + k0 + 32, 0, 0);
      __builtin_prefetch(b1row + k0 + 32, 0, 0);
    }
    acc0 = wmma_bf16(a, b0, acc0);
    acc1 = wmma_bf16(a, b1, acc1);
  }
#pragma unroll
  for (int r = 0; r < 8; ++r) {
    int row = hi16 * 8 + r;
    size_t i0 = (size_t)row * Cn + colBase + m16;
    size_t i1 = (size_t)row * Cn + colBase + 16 + m16;
    au[i0] = logf(fmaxf(acc0[r], 1e-37f)) + (float)pet[i0];
    au[i1] = logf(fmaxf(acc1[r], 1e-37f)) + (float)pet[i1];
  }
}

// per-batch-row LSE, write O[n,t] and renormalized alpha (bf16)
__global__ __launch_bounds__(256) void k_reduce_row(const float* __restrict__ au,
                                                    bf16* __restrict__ alp,
                                                    float* __restrict__ dout,
                                                    const unsigned char* __restrict__ mask,
                                                    int t) {
  __shared__ float red[256];
  int n = blockIdx.x, tid = threadIdx.x;
  const float* row = au + (size_t)n * Cn;
  float m = -3.0e38f;
  for (int c = tid; c < Cn; c += 256) m = fmaxf(m, row[c]);
  red[tid] = m; __syncthreads();
  for (int st = 128; st > 0; st >>= 1) { if (tid < st) red[tid] = fmaxf(red[tid], red[tid + st]); __syncthreads(); }
  m = red[0]; __syncthreads();
  float s = 0.f;
  for (int c = tid; c < Cn; c += 256) s += expf(row[c] - m);
  red[tid] = s; __syncthreads();
  for (int st = 128; st > 0; st >>= 1) { if (tid < st) red[tid] += red[tid + st]; __syncthreads(); }
  float Ot = m + logf(red[0]);
  for (int c = tid; c < Cn; c += 256) alp[(size_t)n * Cn + c] = (bf16)expf(row[c] - Ot);
  if (tid == 0) dout[n * Tn + t] = mask[n * Tn + t] ? Ot : 0.f;
}

// ---------------- host ----------------
static inline int cdiv(long a, long b) { return (int)((a + b - 1) / b); }

extern "C" void kernel_launch(void* const* d_in, const int* in_sizes, int n_in,
                              void* d_out, int out_size, void* d_ws, size_t ws_size,
                              hipStream_t stream) {
  const int* text = (const int*)d_in[0];
  const unsigned char* mask = (const unsigned char*)d_in[1];
  const float* start_emb = (const float*)d_in[2];
  const float* sm_w = (const float*)d_in[3];
  const float* sm_b = (const float*)d_in[4];
  const float* s1w1 = (const float*)d_in[5];
  const float* s1b1 = (const float*)d_in[6];
  const float* s1w2 = (const float*)d_in[7];
  const float* s1b2 = (const float*)d_in[8];
  const float* s2w1 = (const float*)d_in[9];
  const float* s2b1 = (const float*)d_in[10];
  const float* s2w2 = (const float*)d_in[11];
  const float* s2b2 = (const float*)d_in[12];
  const float* t1w1 = (const float*)d_in[13];
  const float* t1b1 = (const float*)d_in[14];
  const float* t1w2 = (const float*)d_in[15];
  const float* t1b2 = (const float*)d_in[16];
  const float* t2w1 = (const float*)d_in[17];
  const float* t2b1 = (const float*)d_in[18];
  const float* t2w2 = (const float*)d_in[19];
  const float* t2b2 = (const float*)d_in[20];
  const float* state_emb = (const float*)d_in[21];
  const float* next_emb = (const float*)d_in[22];
  const float* pret_emb = (const float*)d_in[23];
  const float* term_emb = (const float*)d_in[24];
  const float* proj = (const float*)d_in[25];
  float* dout = (float*)d_out;

  char* base = (char*)d_ws;
  size_t off = 0;
  auto alloc = [&](size_t bytes) -> void* {
    void* p = base + off;
    off += (bytes + 255) & ~(size_t)255;
    return p;
  };

  bf16* projT = (bf16*)alloc((size_t)Dn * Hn * 2);
  bf16* t1w1T = (bf16*)alloc((size_t)Hn * Hn * 2);
  bf16* t1w2T = (bf16*)alloc((size_t)Hn * Hn * 2);
  bf16* t2w1T = (bf16*)alloc((size_t)Hn * Hn * 2);
  bf16* t2w2T = (bf16*)alloc((size_t)Hn * Hn * 2);
  bf16* pt_bf = (bf16*)alloc((size_t)Cn * Hn * 2);
  bf16* st_bf = (bf16*)alloc((size_t)Cn * Hn * 2);
  bf16* nx_bf = (bf16*)alloc((size_t)Cn * Hn * 2);
  bf16* tm_bf = (bf16*)alloc((size_t)Vn * Hn * 2);
  bf16* h_bf = (bf16*)alloc((size_t)Cn * Hn * 2);
  bf16* ft1_bf = (bf16*)alloc((size_t)Cn * Hn * 2);
  bf16* ft_bf = (bf16*)alloc((size_t)Cn * Hn * 2);
  float* st_sq = (float*)alloc((size_t)Cn * 4);
  float* nx_sq = (float*)alloc((size_t)Cn * 4);
  float* tm_sq = (float*)alloc((size_t)Vn * 4);
  float* ft_sq = (float*)alloc((size_t)Cn * 4);
  float* lx = (float*)alloc((size_t)Cn * Dn * 4);
  float* ly = (float*)alloc((size_t)Cn * Dn * 4);
  float* lft = (float*)alloc((size_t)Cn * Dn * 4);
  float* lt = (float*)alloc((size_t)Vn * Dn * 4);
  float* amax_x = (float*)alloc((size_t)Cn * 4);
  float* bmax_y = (float*)alloc((size_t)Cn * 4);
  float* amax_ft = (float*)alloc((size_t)Cn * 4);
  float* bmax_t = (float*)alloc((size_t)Vn * 4);
  bf16* Ax = (bf16*)alloc((size_t)Cn * Dn * 2);
  bf16* By = (bf16*)alloc((size_t)Cn * Dn * 2);
  bf16* Aft = (bf16*)alloc((size_t)Cn * Dn * 2);
  float* logw_y = (float*)alloc((size_t)Dn * 4);
  float* logu_t = (float*)alloc((size_t)Dn * 4);
  float* lden_x = (float*)alloc((size_t)Cn * 4);
  float* lden_ft = (float*)alloc((size_t)Cn * 4);
  bf16* Bg = (bf16*)alloc((size_t)NT * Dn * 2);
  float* bmax_tok = (float*)alloc((size_t)NT * 4);
  bf16* transT = (bf16*)alloc((size_t)Cn * Cn * 2);   // 128 MB, L2-resident
  bf16* pe = (bf16*)alloc((size_t)Tn * Nn_ * Cn * 2); // [T][N][C]
  float* A0 = (float*)alloc((size_t)Dn * 4);
  float* svec = (float*)alloc((size_t)Cn * 4);
  float* lse_s = (float*)alloc(256);
  float* startl = (float*)alloc((size_t)Cn * 4);
  float* au = (float*)alloc((size_t)Nn_ * Cn * 4);
  bf16* alp = (bf16*)alloc((size_t)Nn_ * Cn * 2);

  // ---- weight transposes + bf16 conversions ----
  k_tcvt<<<cdiv((long)Hn * Dn, 256), 256, 0, stream>>>(proj, projT, Hn, Dn);
  k_tcvt<<<cdiv((long)Hn * Hn, 256), 256, 0, stream>>>(t1w1, t1w1T, Hn, Hn);
  k_tcvt<<<cdiv((long)Hn * Hn, 256), 256, 0, stream>>>(t1w2, t1w2T, Hn, Hn);
  k_tcvt<<<cdiv((long)Hn * Hn, 256), 256, 0, stream>>>(t2w1, t2w1T, Hn, Hn);
  k_tcvt<<<cdiv((long)Hn * Hn, 256), 256, 0, stream>>>(t2w2, t2w2T, Hn, Hn);
  k_cvt_bf16<<<cdiv((long)Cn * Hn, 256), 256, 0, stream>>>(pret_emb, pt_bf, (long)Cn * Hn);
  k_cvt_bf16<<<cdiv((long)Cn * Hn, 256), 256, 0, stream>>>(state_emb, st_bf, (long)Cn * Hn);
  k_cvt_bf16<<<cdiv((long)Cn * Hn, 256), 256, 0, stream>>>(next_emb, nx_bf, (long)Cn * Hn);
  k_cvt_bf16<<<cdiv((long)Vn * Hn, 256), 256, 0, stream>>>(term_emb, tm_bf, (long)Vn * Hn);
  k_rowsq_f32<<<Cn, 256, 0, stream>>>(state_emb, st_sq, Hn);
  k_rowsq_f32<<<Cn, 256, 0, stream>>>(next_emb, nx_sq, Hn);
  k_rowsq_f32<<<Vn, 256, 0, stream>>>(term_emb, tm_sq, Hn);

  // ---- start_mlp chain (tiny) ----
  k_start_chain<<<1, 256, 0, stream>>>(start_emb, sm_w, sm_b, s1w1, s1b1, s1w2, s1b2,
                                       s2w1, s2b1, s2w2, s2b2, proj, A0);

  // ---- terminal MLP: two ResLayers via WMMA GEMMs ----
  {
    GArgs g{}; g.M = Cn; g.Nc = Hn; g.K = Hn;
    dim3 grid(cdiv(Hn, 128), cdiv(Cn, 128));
    g.A = pt_bf; g.B = t1w1T; g.outB = h_bf; g.vcol = t1b1;
    k_gemm_nt<0><<<grid, 256, 0, stream>>>(g);
    g.A = h_bf; g.B = t1w2T; g.outB = ft1_bf; g.vcol = t1b2; g.res = pt_bf;
    k_gemm_nt<1><<<grid, 256, 0, stream>>>(g);
    g.A = ft1_bf; g.B = t2w1T; g.outB = h_bf; g.vcol = t2b1; g.res = nullptr;
    k_gemm_nt<0><<<grid, 256, 0, stream>>>(g);
    g.A = h_bf; g.B = t2w2T; g.outB = ft_bf; g.vcol = t2b2; g.res = ft1_bf;
    k_gemm_nt<1><<<grid, 256, 0, stream>>>(g);
  }
  k_rowsq_bf16<<<Cn, 256, 0, stream>>>(ft_bf, ft_sq, Hn);

  // ---- FAVOR+ feature logits: L = X @ proj - 0.5*|x|^2 ----
  {
    GArgs g{}; g.Nc = Dn; g.K = Hn; g.B = projT;
    dim3 gridC(cdiv(Dn, 128), cdiv(Cn, 128));
    g.M = Cn; g.A = st_bf; g.outF = lx; g.vrow = st_sq;
    k_gemm_nt<2><<<gridC, 256, 0, stream>>>(g);
    g.A = nx_bf; g.outF = ly; g.vrow = nx_sq;
    k_gemm_nt<2><<<gridC, 256, 0, stream>>>(g);
    g.A = ft_bf; g.outF = lft; g.vrow = ft_sq;
    k_gemm_nt<2><<<gridC, 256, 0, stream>>>(g);
    dim3 gridV(cdiv(Dn, 128), cdiv(Vn, 128));
    g.M = Vn; g.A = tm_bf; g.outF = lt; g.vrow = tm_sq;
    k_gemm_nt<2><<<gridV, 256, 0, stream>>>(g);
  }

  // ---- row maxes + exp features (bf16), feature-space LSE vectors, denominators ----
  k_rowmax_exp<<<Cn, 256, 0, stream>>>(lx, amax_x, Ax, Dn);
  k_rowmax_exp<<<Cn, 256, 0, stream>>>(ly, bmax_y, By, Dn);
  k_rowmax_exp<<<Cn, 256, 0, stream>>>(lft, amax_ft, Aft, Dn);
  k_rowmax_exp<<<Vn, 256, 0, stream>>>(lt, bmax_t, nullptr, Dn);
  k_collse<<<Dn, 256, 0, stream>>>(ly, logw_y, Cn, Dn);
  k_collse<<<Dn, 256, 0, stream>>>(lt, logu_t, Vn, Dn);
  k_logden<<<Cn, 256, 0, stream>>>(lx, amax_x, logw_y, lden_x, Dn);
  k_logden<<<Cn, 256, 0, stream>>>(lft, amax_ft, logu_t, lden_ft, Dn);
  k_gather<<<cdiv((long)NT * Dn, 256), 256, 0, stream>>>(text, lt, bmax_t, Bg, bmax_tok);

  // ---- transition (stored transposed, bf16) ----
  {
    GArgs g{}; g.M = Cn; g.Nc = Cn; g.K = Dn;
    g.A = By; g.B = Ax; g.outB = transT; g.vrow = bmax_y; g.vcol = lden_x;
    dim3 grid(cdiv(Cn, 128), cdiv(Cn, 128));
    k_gemm_nt<3><<<grid, 256, 0, stream>>>(g);
  }
  // ---- gathered emission log-probs pe[t][n][c] ----
  {
    GArgs g{}; g.M = NT; g.Nc = Cn; g.K = Dn;
    g.A = Bg; g.B = Aft; g.outB = pe; g.vrow = bmax_tok; g.vcol = lden_ft;
    dim3 grid(cdiv(Cn, 128), cdiv(NT, 128));
    k_gemm_nt<4><<<grid, 256, 0, stream>>>(g);
  }

  // ---- start distribution ----
  k_start_gemv<<<Cn / 256, 256, 0, stream>>>(A0, By, bmax_y, svec);
  k_lse_vec<<<1, 256, 0, stream>>>(svec, Cn, lse_s);
  k_sub<<<Cn / 256, 256, 0, stream>>>(svec, lse_s, startl, Cn);

  // ---- scaled forward scan ----
  k_init_au<<<cdiv((long)Nn_ * Cn, 256), 256, 0, stream>>>(startl, pe, au);
  k_reduce_row<<<Nn_, 256, 0, stream>>>(au, alp, dout, mask, 0);
  for (int t = 1; t < Tn; ++t) {
    k_scan_gemm<<<Cn / 256, 256, 0, stream>>>(alp, transT, pe + (size_t)t * Nn_ * Cn, au);
    k_reduce_row<<<Nn_, 256, 0, stream>>>(au, alp, dout, mask, t);
  }
}
